// LightweightMoELayer_21036749816512
// MI455X (gfx1250) — matrix operand order repaired and checked by reading, not compiled, not run on hardware
//
#include <hip/hip_runtime.h>
#include <hip/hip_bf16.h>
#include <math.h>

// ---------------------------------------------------------------------------
// MoE layer for MI455X (gfx1250).
//   T=8192 tokens, D=512, DFF=2048, E=8, top-2 routing.
// Compute-bound (≈2700 FLOP/byte) -> bf16 WMMA path (v_wmma_f32_16x16x32_bf16,
// fp32 accumulation).  32-token tiles so every W fragment feeds two WMMAs.
// X tile staged with GLOBAL_LOAD_ASYNC_TO_LDS_B128 (ASYNCcnt path).
// ---------------------------------------------------------------------------

typedef __attribute__((ext_vector_type(16))) __bf16 v16bf;
typedef __attribute__((ext_vector_type(8)))  float  v8f;

union FragBF {
    v16bf v;
    uint4 q[2];            // two 16-byte chunks (8 bf16 halves each)
};

__device__ __forceinline__ unsigned short f2bf_u16(float f) {
    unsigned u = __builtin_bit_cast(unsigned, f);
    unsigned r = (u + 0x7FFFu + ((u >> 16) & 1u)) >> 16;   // round-nearest-even
    return (unsigned short)r;
}

// ---------------------------------------------------------------------------
// Router: logits = x @ Wr^T ; softmax ; top-2 scatter -> dense probs [T, 8]
// ---------------------------------------------------------------------------
__global__ __launch_bounds__(256)
void router_kernel(const float* __restrict__ x, const float* __restrict__ Wr,
                   float* __restrict__ probs, int T)
{
    const int wave = threadIdx.x >> 5;
    const int lane = threadIdx.x & 31;
    const int t = blockIdx.x * 8 + wave;
    if (t >= T) return;

    const float* xr = x + (size_t)t * 512 + lane * 16;
    float xv[16];
#pragma unroll
    for (int j = 0; j < 16; ++j) xv[j] = xr[j];

    float logit[8];
#pragma unroll
    for (int e = 0; e < 8; ++e) {
        const float* wr = Wr + e * 512 + lane * 16;
        float s = 0.f;
#pragma unroll
        for (int j = 0; j < 16; ++j) s += xv[j] * wr[j];
#pragma unroll
        for (int off = 16; off > 0; off >>= 1)
            s += __shfl_xor(s, off, 32);   // wave32 butterfly reduction
        logit[e] = s;
    }

    float mx = logit[0];
#pragma unroll
    for (int e = 1; e < 8; ++e) mx = fmaxf(mx, logit[e]);
    float sm[8];
    float den = 0.f;
#pragma unroll
    for (int e = 0; e < 8; ++e) { sm[e] = expf(logit[e] - mx); den += sm[e]; }
    float inv = 1.0f / den;
#pragma unroll
    for (int e = 0; e < 8; ++e) sm[e] *= inv;

    int i1 = 0;
#pragma unroll
    for (int e = 1; e < 8; ++e) if (sm[e] > sm[i1]) i1 = e;
    int i2 = -1;
#pragma unroll
    for (int e = 0; e < 8; ++e)
        if (e != i1 && (i2 < 0 || sm[e] > sm[i2])) i2 = e;

    if (lane < 8) {
        float val = 0.f;
#pragma unroll
        for (int e = 0; e < 8; ++e)
            if (lane == e) val = (e == i1 || e == i2) ? sm[e] : 0.f;
        probs[(size_t)t * 8 + lane] = val;
    }
}

// ---------------------------------------------------------------------------
// fp32 -> bf16 conversion (grid-stride)
// ---------------------------------------------------------------------------
__global__ __launch_bounds__(256)
void f32_to_bf16_kernel(const float* __restrict__ src,
                        unsigned short* __restrict__ dst, long n)
{
    long i = (long)blockIdx.x * blockDim.x + threadIdx.x;
    const long stride = (long)gridDim.x * blockDim.x;
    for (; i < n; i += stride) dst[i] = f2bf_u16(src[i]);
}

// ---------------------------------------------------------------------------
// Fused expert FFN, 32-token tiles, 8 waves/block.
//   GEMM1: wave w -> ff cols [w*16, w*16+16), two M-subtiles share B frags
//   GEMM2: wave w -> out cols [w*64, w*64+64), A frags reused across 4 N-tiles
// ---------------------------------------------------------------------------
#define XS_STRIDE 520   // 512 + 8 pad (bf16 elems)
#define HS_STRIDE 136   // 128 + 8 pad

__global__ __launch_bounds__(256)
void moe_ffn_kernel(const unsigned short* __restrict__ x_bf,
                    const unsigned short* __restrict__ w1_bf,
                    const unsigned short* __restrict__ w2_bf,
                    const float* __restrict__ probs,
                    float* __restrict__ out)
{
    __shared__ unsigned short Xs[32 * XS_STRIDE];          // 32.5 KB
    __shared__ unsigned short Hs[2][32 * HS_STRIDE];       // 17   KB
    __shared__ float Ps[32 * 8];                           //  1   KB

    const int tid  = threadIdx.x;
    const int wave = tid >> 5;
    const int lane = tid & 31;
    const int hi   = lane >> 4;     // lane half (K sub-block k = 16*hi + s, this code's order)
    const int ln   = lane & 15;     // A: row M ; B: column N
    const int t0   = blockIdx.x * 32;

    // --- stage X tile [32 x 512] bf16 into LDS via async-to-LDS (ASYNCcnt) ---
    {
        const int row = tid >> 3;                  // 0..31
        const int chk = tid & 7;                   // 0..7, 64 halves (128B) each
        unsigned lds_a = (unsigned)(uintptr_t)&Xs[row * XS_STRIDE + chk * 64];
        unsigned voff  = (unsigned)(((size_t)(t0 + row) * 512 + chk * 64) * 2);
        unsigned long long base = (unsigned long long)(uintptr_t)x_bf;
        asm volatile(
            "global_load_async_to_lds_b128 %0, %1, %2 offset:0\n"
            "global_load_async_to_lds_b128 %0, %1, %2 offset:16\n"
            "global_load_async_to_lds_b128 %0, %1, %2 offset:32\n"
            "global_load_async_to_lds_b128 %0, %1, %2 offset:48\n"
            "global_load_async_to_lds_b128 %0, %1, %2 offset:64\n"
            "global_load_async_to_lds_b128 %0, %1, %2 offset:80\n"
            "global_load_async_to_lds_b128 %0, %1, %2 offset:96\n"
            "global_load_async_to_lds_b128 %0, %1, %2 offset:112\n"
            :: "v"(lds_a), "v"(voff), "s"(base) : "memory");
    }
    Ps[tid] = probs[(size_t)t0 * 8 + tid];         // 256 = 32*8 entries
    asm volatile("s_wait_asynccnt 0x0" ::: "memory");
    __syncthreads();

    v8f yaccL[4], yaccU[4];                        // [rows 0-15], [rows 16-31]
#pragma unroll
    for (int i = 0; i < 4; ++i) { yaccL[i] = {}; yaccU[i] = {}; }

    for (int e = 0; e < 8; ++e) {
        // skip experts routed to none of the 32 tokens (exact: weight==0)
        float pm = 0.f;
#pragma unroll
        for (int r = 0; r < 32; ++r) pm = fmaxf(pm, Ps[r * 8 + e]);
        if (pm == 0.f) continue;

        const unsigned short* w1e = w1_bf + (size_t)e * 2048 * 512;
        const unsigned short* w2e = w2_bf + (size_t)e * 512 * 2048;

        for (int it = 0; it < 16; ++it) {
            const int ff0 = it * 128;
            const int buf = it & 1;

            // ---- GEMM1: H[0:32, ff0 + wave*16 + ln], K = 512 ----
            v8f hacc0 = {}, hacc1 = {};
            const int bcol = ff0 + wave * 16 + ln;                 // W1 row
            const unsigned short* bptr  = w1e + (size_t)bcol * 512 + hi * 16;
            const unsigned short* arowL = &Xs[ ln       * XS_STRIDE + hi * 16];
            const unsigned short* arowU = &Xs[(16 + ln) * XS_STRIDE + hi * 16];
            __builtin_prefetch(w1e + (size_t)(bcol + 128) * 512 + hi * 16, 0, 1);
#pragma unroll
            for (int k0 = 0; k0 < 512; k0 += 32) {
                FragBF aL, aU, b;
                b.q[0]  = *(const uint4*)(bptr + k0);       // K = k0+hi*16 ..
                b.q[1]  = *(const uint4*)(bptr + k0 + 8);
                aL.q[0] = *(const uint4*)(arowL + k0);      // K = k0+hi*16 ..
                aL.q[1] = *(const uint4*)(arowL + k0 + 8);
                aU.q[0] = *(const uint4*)(arowU + k0);
                aU.q[1] = *(const uint4*)(arowU + k0 + 8);
                hacc0 = __builtin_amdgcn_wmma_f32_16x16x32_bf16(
                            false, aL.v, false, b.v, (short)0, hacc0, false, false);
                hacc1 = __builtin_amdgcn_wmma_f32_16x16x32_bf16(
                            false, aU.v, false, b.v, (short)0, hacc1, false, false);
            }

            // ---- exact-erf GELU, fold router weight, pack bf16 to LDS ----
#pragma unroll
            for (int r = 0; r < 8; ++r) {
                const int M0 = r + hi * 8;                  // C layout: row
                const int M1 = 16 + M0;
                float h0 = hacc0[r], h1 = hacc1[r];
                float g0 = 0.5f * h0 * (1.0f + erff(h0 * 0.70710678118654752f));
                float g1 = 0.5f * h1 * (1.0f + erff(h1 * 0.70710678118654752f));
                Hs[buf][M0 * HS_STRIDE + wave * 16 + ln] =
                    f2bf_u16(g0 * Ps[M0 * 8 + e]);
                Hs[buf][M1 * HS_STRIDE + wave * 16 + ln] =
                    f2bf_u16(g1 * Ps[M1 * 8 + e]);
            }
            __syncthreads();   // one barrier per iter (Hs double-buffered)

            // ---- GEMM2: Y[0:32, wave*64 .. +63] += H[32x128] @ W2^T ----
            const unsigned short* arow2L = &Hs[buf][ ln       * HS_STRIDE + hi * 16];
            const unsigned short* arow2U = &Hs[buf][(16 + ln) * HS_STRIDE + hi * 16];
#pragma unroll
            for (int k0 = 0; k0 < 128; k0 += 32) {
                FragBF aL, aU;
                aL.q[0] = *(const uint4*)(arow2L + k0);
                aL.q[1] = *(const uint4*)(arow2L + k0 + 8);
                aU.q[0] = *(const uint4*)(arow2U + k0);
                aU.q[1] = *(const uint4*)(arow2U + k0 + 8);
#pragma unroll
                for (int nt = 0; nt < 4; ++nt) {
                    const int col = wave * 64 + nt * 16 + ln;   // out dim
                    const unsigned short* bptr2 =
                        w2e + (size_t)col * 2048 + ff0 + k0 + hi * 16;
                    FragBF b;
                    b.q[0] = *(const uint4*)(bptr2);
                    b.q[1] = *(const uint4*)(bptr2 + 8);
                    yaccL[nt] = __builtin_amdgcn_wmma_f32_16x16x32_bf16(
                                    false, aL.v, false, b.v, (short)0, yaccL[nt],
                                    false, false);
                    yaccU[nt] = __builtin_amdgcn_wmma_f32_16x16x32_bf16(
                                    false, aU.v, false, b.v, (short)0, yaccU[nt],
                                    false, false);
                }
            }
        }
    }

    // ---- write out [32 x 512] fp32; each element owned by exactly one lane
#pragma unroll
    for (int nt = 0; nt < 4; ++nt)
#pragma unroll
        for (int r = 0; r < 8; ++r) {
            const int M = r + hi * 8;
            const size_t c = wave * 64 + nt * 16 + ln;
            out[(size_t)(t0 + M)      * 512 + c] = yaccL[nt][r];
            out[(size_t)(t0 + 16 + M) * 512 + c] = yaccU[nt][r];
        }
}

// ---------------------------------------------------------------------------
extern "C" void kernel_launch(void* const* d_in, const int* in_sizes, int n_in,
                              void* d_out, int out_size, void* d_ws, size_t ws_size,
                              hipStream_t stream) {
    const float* x  = (const float*)d_in[0];   // [4,2048,512]
    const float* Wr = (const float*)d_in[1];   // [8,512]
    const float* W1 = (const float*)d_in[2];   // [8,2048,512]
    const float* W2 = (const float*)d_in[3];   // [8,512,2048]
    float* out = (float*)d_out;

    const int D = 512, E = 8, DFF = 2048;
    const long T = (long)in_sizes[0] / D;      // 8192

    // workspace carve-up (all chunks are multiples of 256 bytes): ~42.2 MB
    char* ws = (char*)d_ws;
    size_t off = 0;
    float* probs = (float*)(ws + off);                   off += (size_t)T * E * 4;
    unsigned short* x_bf  = (unsigned short*)(ws + off); off += (size_t)T * D * 2;
    unsigned short* w1_bf = (unsigned short*)(ws + off); off += (size_t)E * DFF * D * 2;
    unsigned short* w2_bf = (unsigned short*)(ws + off); off += (size_t)E * D * DFF * 2;
    (void)ws_size; (void)n_in; (void)out_size;

    router_kernel<<<dim3((unsigned)((T + 7) / 8)), dim3(256), 0, stream>>>(
        x, Wr, probs, (int)T);

    f32_to_bf16_kernel<<<dim3(1024), dim3(256), 0, stream>>>(x, x_bf, T * D);
    f32_to_bf16_kernel<<<dim3(2048), dim3(256), 0, stream>>>(
        W1, w1_bf, (long)E * DFF * D);
    f32_to_bf16_kernel<<<dim3(2048), dim3(256), 0, stream>>>(
        W2, w2_bf, (long)E * D * DFF);

    moe_ffn_kernel<<<dim3((unsigned)(T / 32)), dim3(256), 0, stream>>>(
        x_bf, w1_bf, w2_bf, probs, out);
}
